// ViT_50732153701015
// MI455X (gfx1250) — compile-verified, matrix operand order checked by reading, not run on hardware
//
#include <hip/hip_runtime.h>
#include <hip/hip_bf16.h>

// ---------------------------------------------------------------------------
// Types
// ---------------------------------------------------------------------------
using half_t = _Float16;
typedef _Float16     v16h  __attribute__((ext_vector_type(16)));
typedef float        v8f   __attribute__((ext_vector_type(8)));
typedef unsigned int u32x4 __attribute__((ext_vector_type(4)));
typedef float        f32x4 __attribute__((ext_vector_type(4)));

// ---------------------------------------------------------------------------
// Problem constants (fixed by the reference)
// ---------------------------------------------------------------------------
constexpr int BATCH = 4;
constexpr int NTOK  = 256;
constexpr int DIM   = 1536;
constexpr int HEADS = 12;
constexpr int DH    = 64;
constexpr int INNER = HEADS * DH;     // 768
constexpr int QKV   = 3 * INNER;      // 2304
constexpr int MLP   = 6144;
constexpr int DEPTH = 4;
constexpr int MROWS = BATCH * NTOK;   // 1024

// ---------------------------------------------------------------------------
// Helpers
// ---------------------------------------------------------------------------
static __device__ inline v8f vzero8() {
  v8f z;
#pragma unroll
  for (int i = 0; i < 8; ++i) z[i] = 0.0f;
  return z;
}

// CDNA5 WMMA f16 A/B fragment: lane holds one row(col); halves
// [koff..koff+7] and [koff+16..koff+23] of a 32-wide K tile.
static __device__ inline v16h ld_frag(const half_t* p) {
  union { v16h h; u32x4 u[2]; } f;
  f.u[0] = *(const u32x4*)(p);
  f.u[1] = *(const u32x4*)(p + 16);
  return f.h;
}

static __device__ inline float gelu_tanh(float x) {
  float u = 0.7978845608028654f * (x + 0.044715f * x * x * x);
  float e = __expf(-2.0f * u);
  float th = 2.0f / (1.0f + e) - 1.0f;
  return 0.5f * x * (1.0f + th);
}

// Async global->LDS copy of 16B per lane (GVS mode: SGPR64 base + VGPR32 off).
// lds = low 32 bits of flat shared addr (== LDS byte offset per aperture map).
static __device__ inline void async_b128(unsigned lds, unsigned goff,
                                         const void* base) {
  asm volatile("global_load_async_to_lds_b128 %0, %1, %2"
               :
               : "v"(lds), "v"(goff), "s"(base)
               : "memory");
}
static __device__ inline unsigned lds_addr(const void* p) {
  return (unsigned)(uintptr_t)p;
}

// ---------------------------------------------------------------------------
// Patchify: x[B,3,8,128,128] -> tokens[B*N, 1536]
// ---------------------------------------------------------------------------
__global__ __launch_bounds__(256) void patchify_k(const float* __restrict__ X,
                                                  float* __restrict__ T) {
  const int tid = threadIdx.x;
  const int n = blockIdx.x % NTOK;
  const int b = blockIdx.x / NTOK;
  const int f = n >> 6, rem = n & 63, hh = rem >> 3, ww = rem & 7;
#pragma unroll
  for (int i = 0; i < 6; ++i) {
    const int dim = tid + i * 256;
    const int p1 = dim / 96;
    const int rr = dim % 96;
    const int p2 = rr / 6;
    const int r2 = rr % 6;
    const int pf = r2 / 3;
    const int c  = r2 % 3;
    const int Fi = f * 2 + pf, Hi = hh * 16 + p1, Wi = ww * 16 + p2;
    T[(size_t)blockIdx.x * DIM + dim] =
        X[(((size_t)(b * 3 + c) * 8 + Fi) * 128 + Hi) * 128 + Wi];
  }
}

// ---------------------------------------------------------------------------
// Unpatchify: tokens[B*N,1536] -> out[B,3,8,128,128]
// ---------------------------------------------------------------------------
__global__ __launch_bounds__(256) void unpatchify_k(const float* __restrict__ T,
                                                    float* __restrict__ O) {
  int idx = blockIdx.x * 256 + threadIdx.x;
  const int Wi = idx & 127;
  int t2 = idx >> 7;
  const int Hi = t2 & 127; t2 >>= 7;
  const int Fi = t2 & 7;   t2 >>= 3;
  const int c  = t2 % 3;
  const int b  = t2 / 3;
  const int f = Fi >> 1, pf = Fi & 1;
  const int hh = Hi >> 4, p1 = Hi & 15;
  const int ww = Wi >> 4, p2 = Wi & 15;
  const int n   = f * 64 + hh * 8 + ww;
  const int dim = p1 * 96 + p2 * 6 + pf * 3 + c;
  O[idx] = T[(size_t)(b * NTOK + n) * DIM + dim];
}

// ---------------------------------------------------------------------------
// Weight convert + transpose: W[K][N] f32 -> Bt[N][K] f16 (64x64 LDS tiles)
// ---------------------------------------------------------------------------
__global__ __launch_bounds__(256) void wcvt_k(const float* __restrict__ W,
                                              half_t* __restrict__ Bt,
                                              int K, int N) {
  __shared__ float sh[64][68];
  const int tid = threadIdx.x;
  const int kT = blockIdx.y * 64;
  const int nT = blockIdx.x * 64;
  const int rr = tid >> 2;
  const int cc = (tid & 3) * 16;
  const float* src = W + (size_t)(kT + rr) * N + nT + cc;
#pragma unroll
  for (int s4 = 0; s4 < 4; ++s4) {
    f32x4 v = *(const f32x4*)(src + s4 * 4);
#pragma unroll
    for (int j = 0; j < 4; ++j) sh[rr][cc + s4 * 4 + j] = v[j];
  }
  __syncthreads();
  half_t* dst = Bt + (size_t)(nT + rr) * K + kT + cc;
  union { u32x4 u; half_t h[8]; } o0, o1;
#pragma unroll
  for (int j = 0; j < 8; ++j) {
    o0.h[j] = (half_t)sh[cc + j][rr];
    o1.h[j] = (half_t)sh[cc + 8 + j][rr];
  }
  *(u32x4*)(dst)     = o0.u;
  *(u32x4*)(dst + 8) = o1.u;
}

// ---------------------------------------------------------------------------
// Row LayerNorm over 1536 cols; optional pos-emb add; OutT = f16 or f32.
// ---------------------------------------------------------------------------
template <typename OutT, bool ADDPOS>
__global__ __launch_bounds__(256) void layernorm_k(const float* __restrict__ X,
                                                   const float* __restrict__ g,
                                                   const float* __restrict__ bta,
                                                   OutT* __restrict__ Y,
                                                   const float* __restrict__ pos) {
  __shared__ float s1[256];
  __shared__ float s2[256];
  const int row = blockIdx.x, tid = threadIdx.x;
  const float* xr = X + (size_t)row * DIM;
  float vals[6];
  float ls = 0.0f, lq = 0.0f;
#pragma unroll
  for (int i = 0; i < 6; ++i) {
    float v = xr[tid + i * 256];
    vals[i] = v;
    ls += v;
    lq += v * v;
  }
  s1[tid] = ls;
  s2[tid] = lq;
  __syncthreads();
  for (int off = 128; off > 0; off >>= 1) {
    if (tid < off) { s1[tid] += s1[tid + off]; s2[tid] += s2[tid + off]; }
    __syncthreads();
  }
  const float mean = s1[0] * (1.0f / 1536.0f);
  const float var  = s2[0] * (1.0f / 1536.0f) - mean * mean;
  const float rstd = rsqrtf(var + 1e-5f);
  const float* pr = ADDPOS ? (pos + (size_t)(row % NTOK) * DIM) : nullptr;
#pragma unroll
  for (int i = 0; i < 6; ++i) {
    const int c = tid + i * 256;
    float y = (vals[i] - mean) * rstd * g[c] + bta[c];
    if (ADDPOS) y += pr[c];
    Y[(size_t)row * DIM + c] = (OutT)y;
  }
}

// ---------------------------------------------------------------------------
// WMMA GEMM:  C[M,N] = A[M,K](f16) * Bt[N,K](f16) (+bias)(gelu?)(+=resid?)
// 128x128 block, BK=32, double-buffered LDS fed by async global->LDS b128,
// software-pipelined with s_wait_asynccnt. 8 waves (2x4), wave = 64x32.
// ---------------------------------------------------------------------------
template <bool BIAS, bool GELU, bool RESID, bool HOUT>
__global__ __launch_bounds__(256) void gemm_f16_wmma(
    const half_t* __restrict__ A, const half_t* __restrict__ Bt,
    const float* __restrict__ bias, float* __restrict__ Cf,
    half_t* __restrict__ Ch, int M, int N, int K) {
  constexpr int LDA  = 40;              // halves; %8==0 -> 16B aligned
  constexpr int TILE = 128 * LDA;
  __shared__ half_t As[2 * TILE];
  __shared__ half_t Bs[2 * TILE];

  const int tid  = threadIdx.x;
  const int lane = tid & 31;
  const int w    = tid >> 5;
  const int wm   = w & 1;
  const int wn   = w >> 1;
  const int r    = lane & 15;
  const int hi   = lane >> 4;
  const int koff = hi * 8;
  const int rowB = blockIdx.y * 128;
  const int colB = blockIdx.x * 128;

  v8f acc[4][2];
#pragma unroll
  for (int mt = 0; mt < 4; ++mt)
#pragma unroll
    for (int nt = 0; nt < 2; ++nt) acc[mt][nt] = vzero8();

  // staging map: thread -> (row, 8-half segment), rows r0 and r0+64
  const int srow = tid >> 2;
  const int sseg = (tid & 3) * 8;

  unsigned ldsA[2][2], ldsB[2][2];
#pragma unroll
  for (int bf = 0; bf < 2; ++bf)
#pragma unroll
    for (int j = 0; j < 2; ++j) {
      ldsA[bf][j] = lds_addr(&As[bf * TILE + (srow + j * 64) * LDA + sseg]);
      ldsB[bf][j] = lds_addr(&Bs[bf * TILE + (srow + j * 64) * LDA + sseg]);
    }
  unsigned voffA[2], voffB[2];
#pragma unroll
  for (int j = 0; j < 2; ++j) {
    voffA[j] = (unsigned)(((size_t)(rowB + srow + j * 64) * K + sseg) * 2);
    voffB[j] = (unsigned)(((size_t)(colB + srow + j * 64) * K + sseg) * 2);
  }

  auto issue = [&](int bf) {
#pragma unroll
    for (int j = 0; j < 2; ++j) {
      async_b128(ldsA[bf][j], voffA[j], A);
      async_b128(ldsB[bf][j], voffB[j], Bt);
      voffA[j] += 64;                    // advance one 32-half K tile
      voffB[j] += 64;
    }
  };

  const int NT = K >> 5;
  issue(0);
  for (int kt = 0; kt < NT; ++kt) {
    const int buf = kt & 1;
    if (kt + 1 < NT) {
      issue(buf ^ 1);
      asm volatile("s_wait_asynccnt 0x4" ::: "memory");  // tile kt landed
    } else {
      asm volatile("s_wait_asynccnt 0x0" ::: "memory");
    }
    __syncthreads();

    const half_t* ab = As + buf * TILE;
    const half_t* bb = Bs + buf * TILE;
    v16h af[4], bfv[2];
#pragma unroll
    for (int mt = 0; mt < 4; ++mt)
      af[mt] = ld_frag(ab + (wm * 64 + mt * 16 + r) * LDA + koff);
#pragma unroll
    for (int nt = 0; nt < 2; ++nt)
      bfv[nt] = ld_frag(bb + (wn * 32 + nt * 16 + r) * LDA + koff);
#pragma unroll
    for (int mt = 0; mt < 4; ++mt)
#pragma unroll
      for (int nt = 0; nt < 2; ++nt)
        acc[mt][nt] = __builtin_amdgcn_wmma_f32_16x16x32_f16(
            false, af[mt], false, bfv[nt], (short)0, acc[mt][nt], false, false);
    __syncthreads();
  }

  // epilogue
#pragma unroll
  for (int mt = 0; mt < 4; ++mt) {
#pragma unroll
    for (int nt = 0; nt < 2; ++nt) {
      const int col = colB + wn * 32 + nt * 16 + r;
      const float bb2 = BIAS ? bias[col] : 0.0f;
#pragma unroll
      for (int i = 0; i < 8; ++i) {
        const int row = rowB + wm * 64 + mt * 16 + i + 8 * hi;
        float v = acc[mt][nt][i] + bb2;
        if (GELU) v = gelu_tanh(v);
        const size_t idx = (size_t)row * N + col;
        if (HOUT)        Ch[idx] = (half_t)v;
        else if (RESID)  Cf[idx] += v;
        else             Cf[idx] = v;
      }
    }
  }
}

// ---------------------------------------------------------------------------
// Flash-style cross attention per (batch, head, 128-row block).
// K chunk staged by async global->LDS; V transposed through VGPRs.
// ---------------------------------------------------------------------------
__global__ __launch_bounds__(256) void attn_wmma(const half_t* __restrict__ Qsrc,
                                                 const half_t* __restrict__ KVsrc,
                                                 half_t* __restrict__ O) {
  __shared__ half_t Ks[64 * 72];          // [key][d]
  __shared__ half_t Vt[64 * 72];          // [d][key]
  __shared__ half_t Pb[8 * 16 * 72];      // per-wave P rows

  const int tid = threadIdx.x;
  const int lane = tid & 31;
  const int w = tid >> 5;
  const int r = lane & 15;
  const int hi = lane >> 4;
  const int koff = hi * 8;

  int bid = blockIdx.x;
  const int rb = bid & 1; bid >>= 1;
  const int h = bid % HEADS;
  const int b = bid / HEADS;
  const int rowbase = rb * 128;
  const float scale = 0.125f;             // DH^-0.5

  const half_t* qrow =
      Qsrc + (size_t)(b * NTOK + rowbase + w * 16 + r) * QKV + h * DH;
  v16h aq[2];
  aq[0] = ld_frag(qrow + koff);
  aq[1] = ld_frag(qrow + 32 + koff);

  float mrun[8], lrun[8];
  v8f o[4];
#pragma unroll
  for (int i = 0; i < 8; ++i) { mrun[i] = -3e38f; lrun[i] = 0.0f; }
#pragma unroll
  for (int jd = 0; jd < 4; ++jd) o[jd] = vzero8();

  const int skey = tid >> 2;
  const int ss   = (tid & 3) * 16;
  const unsigned ldsK0 = lds_addr(&Ks[skey * 72 + ss]);
  const unsigned ldsK1 = lds_addr(&Ks[skey * 72 + ss + 8]);

  for (int jc = 0; jc < 4; ++jc) {
    // ---- stage K chunk (async) and V chunk (transposed via VGPRs) ----
    {
      const size_t krow = (size_t)(b * NTOK + jc * 64 + skey) * QKV;
      const unsigned voffK =
          (unsigned)((krow + INNER + h * DH + ss) * 2);
      async_b128(ldsK0, voffK, KVsrc);
      async_b128(ldsK1, voffK + 16, KVsrc);

      const half_t* gv = KVsrc + krow + 2 * INNER + h * DH + ss;
      union { u32x4 u; half_t hh[8]; } v0, v1;
      v0.u = *(const u32x4*)gv;
      v1.u = *(const u32x4*)(gv + 8);
#pragma unroll
      for (int j = 0; j < 8; ++j) {
        Vt[(ss + j) * 72 + skey]     = v0.hh[j];
        Vt[(ss + 8 + j) * 72 + skey] = v1.hh[j];
      }
    }
    asm volatile("s_wait_asynccnt 0x0" ::: "memory");
    __syncthreads();

    // ---- S = Q * K^T ----
    v8f s[4];
#pragma unroll
    for (int jn = 0; jn < 4; ++jn) {
      v8f sc = vzero8();
#pragma unroll
      for (int kk = 0; kk < 2; ++kk) {
        v16h bK = ld_frag(Ks + (jn * 16 + r) * 72 + kk * 32 + koff);
        sc = __builtin_amdgcn_wmma_f32_16x16x32_f16(
            false, aq[kk], false, bK, (short)0, sc, false, false);
      }
      s[jn] = sc;
    }

    // ---- online softmax ----
    float fac[8];
#pragma unroll
    for (int i = 0; i < 8; ++i) {
      float m = -3e38f;
#pragma unroll
      for (int jn = 0; jn < 4; ++jn) {
        s[jn][i] *= scale;
        m = fmaxf(m, s[jn][i]);
      }
      m = fmaxf(m, __shfl_xor(m, 1, 16));
      m = fmaxf(m, __shfl_xor(m, 2, 16));
      m = fmaxf(m, __shfl_xor(m, 4, 16));
      m = fmaxf(m, __shfl_xor(m, 8, 16));
      const float mn = fmaxf(mrun[i], m);
      fac[i] = __expf(mrun[i] - mn);
      mrun[i] = mn;
      float rs = 0.0f;
#pragma unroll
      for (int jn = 0; jn < 4; ++jn) {
        float pv = __expf(s[jn][i] - mn);
        s[jn][i] = pv;
        rs += pv;
      }
      rs += __shfl_xor(rs, 1, 16);
      rs += __shfl_xor(rs, 2, 16);
      rs += __shfl_xor(rs, 4, 16);
      rs += __shfl_xor(rs, 8, 16);
      lrun[i] = lrun[i] * fac[i] + rs;
    }
#pragma unroll
    for (int jd = 0; jd < 4; ++jd)
#pragma unroll
      for (int i = 0; i < 8; ++i) o[jd][i] *= fac[i];

    // ---- bounce P through LDS (C-frag -> A-frag relayout) ----
    half_t* pw = Pb + w * 16 * 72;
#pragma unroll
    for (int jn = 0; jn < 4; ++jn)
#pragma unroll
      for (int i = 0; i < 8; ++i)
        pw[(i + 8 * hi) * 72 + jn * 16 + r] = (half_t)s[jn][i];
    __syncthreads();

    // ---- O += P * V ----
    v16h ap[2];
    ap[0] = ld_frag(pw + r * 72 + koff);
    ap[1] = ld_frag(pw + r * 72 + 32 + koff);
#pragma unroll
    for (int jd = 0; jd < 4; ++jd) {
#pragma unroll
      for (int kk = 0; kk < 2; ++kk) {
        v16h bv = ld_frag(Vt + (jd * 16 + r) * 72 + kk * 32 + koff);
        o[jd] = __builtin_amdgcn_wmma_f32_16x16x32_f16(
            false, ap[kk], false, bv, (short)0, o[jd], false, false);
      }
    }
    __syncthreads();
  }

  float inv[8];
#pragma unroll
  for (int i = 0; i < 8; ++i) inv[i] = 1.0f / lrun[i];
#pragma unroll
  for (int jd = 0; jd < 4; ++jd)
#pragma unroll
    for (int i = 0; i < 8; ++i) {
      const size_t row = (size_t)(b * NTOK + rowbase + w * 16 + i + 8 * hi);
      O[row * INNER + h * DH + jd * 16 + r] = (half_t)(o[jd][i] * inv[i]);
    }
}

// ---------------------------------------------------------------------------
// Host orchestration
// ---------------------------------------------------------------------------
extern "C" void kernel_launch(void* const* d_in, const int* in_sizes, int n_in,
                              void* d_out, int out_size, void* d_ws, size_t ws_size,
                              hipStream_t stream) {
  const float* X       = (const float*)d_in[0];
  const float* X1      = (const float*)d_in[1];
  const float* ln_p1_g = (const float*)d_in[2];
  const float* ln_p1_b = (const float*)d_in[3];
  const float* W_patch = (const float*)d_in[4];
  const float* b_patch = (const float*)d_in[5];
  const float* ln_p2_g = (const float*)d_in[6];
  const float* ln_p2_b = (const float*)d_in[7];
  const float* pos_emb = (const float*)d_in[8];
  const float* ln_a_g  = (const float*)d_in[9];
  const float* ln_a_b  = (const float*)d_in[10];
  const float* W_qkv   = (const float*)d_in[11];
  const float* W_o     = (const float*)d_in[12];
  const float* b_o     = (const float*)d_in[13];
  const float* ln_f_g  = (const float*)d_in[14];
  const float* ln_f_b  = (const float*)d_in[15];
  const float* W_f1    = (const float*)d_in[16];
  const float* b_f1    = (const float*)d_in[17];
  const float* W_f2    = (const float*)d_in[18];
  const float* b_f2    = (const float*)d_in[19];
  const float* ln_f1_g = (const float*)d_in[20];
  const float* ln_f1_b = (const float*)d_in[21];
  const float* W_f1_1  = (const float*)d_in[22];
  const float* b_f1_1  = (const float*)d_in[23];
  const float* W_f2_1  = (const float*)d_in[24];
  const float* b_f2_1  = (const float*)d_in[25];
  float* out = (float*)d_out;

  // ---- workspace carve ----
  char* p = (char*)d_ws;
  auto carve = [&](size_t bytes) {
    void* q = (void*)p;
    p += (bytes + 255) & ~(size_t)255;
    return q;
  };
  float*  t0   = (float*)carve((size_t)MROWS * DIM * 4);
  float*  t1   = (float*)carve((size_t)MROWS * DIM * 4);
  float*  tok  = (float*)carve((size_t)MROWS * DIM * 4);
  half_t* lnh  = (half_t*)carve((size_t)MROWS * DIM * 2);
  half_t* qkv0 = (half_t*)carve((size_t)MROWS * QKV * 2);
  half_t* qkv1 = (half_t*)carve((size_t)MROWS * QKV * 2);
  half_t* o0   = (half_t*)carve((size_t)MROWS * INNER * 2);
  half_t* o1   = (half_t*)carve((size_t)MROWS * INNER * 2);
  half_t* hmlp = (half_t*)carve((size_t)MROWS * MLP * 2);
  half_t* wbuf = (half_t*)carve((size_t)DIM * MLP * 2);   // f16 [N][K] weights
  (void)ws_size; (void)n_in; (void)in_sizes; (void)out_size;

  const dim3 blk(256);
  const dim3 gDIM(DIM / 128, MROWS / 128);
  const dim3 gQKV(QKV / 128, MROWS / 128);
  const dim3 gMLP(MLP / 128, MROWS / 128);

  auto cvt = [&](const float* W, int K, int N) {
    wcvt_k<<<dim3(N / 64, K / 64), blk, 0, stream>>>(W, wbuf, K, N);
  };

  // ---- patch embed (both streams share converted W_patch) ----
  cvt(W_patch, DIM, DIM);
  auto embed = [&](const float* x, float* t) {
    patchify_k<<<MROWS, blk, 0, stream>>>(x, tok);
    layernorm_k<half_t, false><<<MROWS, blk, 0, stream>>>(tok, ln_p1_g, ln_p1_b,
                                                          lnh, nullptr);
    gemm_f16_wmma<true, false, false, false><<<gDIM, blk, 0, stream>>>(
        lnh, wbuf, b_patch, t, nullptr, MROWS, DIM, DIM);
    layernorm_k<float, true><<<MROWS, blk, 0, stream>>>(t, ln_p2_g, ln_p2_b, t,
                                                        pos_emb);
  };
  embed(X, t0);
  embed(X1, t1);

  // ---- transformer layers ----
  for (int d = 0; d < DEPTH; ++d) {
    const float* ag = ln_a_g + d * DIM;
    const float* ab = ln_a_b + d * DIM;
    const float* bo = b_o + d * DIM;

    // QKV (shared converted weight, both streams)
    cvt(W_qkv + (size_t)d * DIM * QKV, DIM, QKV);
    layernorm_k<half_t, false><<<MROWS, blk, 0, stream>>>(t0, ag, ab, lnh, nullptr);
    gemm_f16_wmma<false, false, false, true><<<gQKV, blk, 0, stream>>>(
        lnh, wbuf, nullptr, nullptr, qkv0, MROWS, QKV, DIM);
    layernorm_k<half_t, false><<<MROWS, blk, 0, stream>>>(t1, ag, ab, lnh, nullptr);
    gemm_f16_wmma<false, false, false, true><<<gQKV, blk, 0, stream>>>(
        lnh, wbuf, nullptr, nullptr, qkv1, MROWS, QKV, DIM);

    // cross attention
    attn_wmma<<<BATCH * HEADS * 2, blk, 0, stream>>>(qkv0, qkv1, o0);
    attn_wmma<<<BATCH * HEADS * 2, blk, 0, stream>>>(qkv1, qkv0, o1);

    // output projection (+residual)
    cvt(W_o + (size_t)d * INNER * DIM, INNER, DIM);
    gemm_f16_wmma<true, false, true, false><<<gDIM, blk, 0, stream>>>(
        o0, wbuf, bo, t0, nullptr, MROWS, DIM, INNER);
    gemm_f16_wmma<true, false, true, false><<<gDIM, blk, 0, stream>>>(
        o1, wbuf, bo, t1, nullptr, MROWS, DIM, INNER);

    // FFN stream 0
    layernorm_k<half_t, false><<<MROWS, blk, 0, stream>>>(
        t0, ln_f_g + d * DIM, ln_f_b + d * DIM, lnh, nullptr);
    cvt(W_f1 + (size_t)d * DIM * MLP, DIM, MLP);
    gemm_f16_wmma<true, true, false, true><<<gMLP, blk, 0, stream>>>(
        lnh, wbuf, b_f1 + d * MLP, nullptr, hmlp, MROWS, MLP, DIM);
    cvt(W_f2 + (size_t)d * MLP * DIM, MLP, DIM);
    gemm_f16_wmma<true, false, true, false><<<gDIM, blk, 0, stream>>>(
        hmlp, wbuf, b_f2 + d * DIM, t0, nullptr, MROWS, DIM, MLP);

    // FFN stream 1
    layernorm_k<half_t, false><<<MROWS, blk, 0, stream>>>(
        t1, ln_f1_g + d * DIM, ln_f1_b + d * DIM, lnh, nullptr);
    cvt(W_f1_1 + (size_t)d * DIM * MLP, DIM, MLP);
    gemm_f16_wmma<true, true, false, true><<<gMLP, blk, 0, stream>>>(
        lnh, wbuf, b_f1_1 + d * MLP, nullptr, hmlp, MROWS, MLP, DIM);
    cvt(W_f2_1 + (size_t)d * MLP * DIM, MLP, DIM);
    gemm_f16_wmma<true, false, true, false><<<gDIM, blk, 0, stream>>>(
        hmlp, wbuf, b_f2_1 + d * DIM, t1, nullptr, MROWS, DIM, MLP);
  }

  // ---- unpatchify both streams into d_out ----
  constexpr int OUTELEMS = BATCH * 3 * 8 * 128 * 128;
  unpatchify_k<<<OUTELEMS / 256, blk, 0, stream>>>(t0, out);
  unpatchify_k<<<OUTELEMS / 256, blk, 0, stream>>>(t1, out + OUTELEMS);
}
